// GAT_batchnorm_75479755259984
// MI455X (gfx1250) — matrix operand order repaired and checked by reading, not compiled, not run on hardware
//
#include <hip/hip_runtime.h>
#include <hip/hip_bf16.h>
#include <math.h>

// Problem constants (from reference)
#define NNODES   50000
#define NEDGES   800000
#define FIN      128
#define HEADS    8
#define CH       16
#define HCF      128      // HEADS*CH
#define OUTF     16
#define NEG_SLOPE 0.2f

typedef __attribute__((ext_vector_type(16))) _Float16 v16h;
typedef __attribute__((ext_vector_type(8)))  _Float16 v8h;
typedef __attribute__((ext_vector_type(8)))  float    v8f;

// ---------------------------------------------------------------------------
// f32 -> f16 elementwise convert (feeds WMMA)
// ---------------------------------------------------------------------------
__global__ void k_f32_to_f16(const float* __restrict__ in,
                             _Float16* __restrict__ out, int n) {
  int i = blockIdx.x * blockDim.x + threadIdx.x;
  if (i < n) out[i] = (_Float16)in[i];
}

// ---------------------------------------------------------------------------
// GEMM C[M, NT*16] = A[M,K] (f16, row-major) x B[K, NT*16] (f16, row-major),
// f32 accumulate.  One wave32 owns a 16-row block and ALL NT column tiles:
// the A fragment is loaded once per K-step and reused for NT WMMAs
// (8x fewer A loads than one-tile-per-wave; W stays hot in WGP$/L2).
//
// A lane layout (16-bit A 16x32): lane l -> row M=l&15; g=l>>4;
//   elements 0..7  = K in [k0+8g,   k0+8g+7]   (contiguous 16B load)
//   elements 8..15 = K in [k0+16+8g,k0+16+8g+7](contiguous 16B load)
// B lane layout (32x16): lane l -> col N=l&15; elements follow A's K order (strided gather).
// C layout: VGPR r, lanes 0-15: C[M=r][N=lane]; lanes 16-31: C[M=r+8][N=lane-16].
// ---------------------------------------------------------------------------
template <int NT>
__global__ void k_gemm_wmma(const _Float16* __restrict__ A,
                            const _Float16* __restrict__ B,
                            float* __restrict__ C,
                            int M, int K) {
  const int Nc   = NT * 16;
  const int lane = threadIdx.x & 31;
  const int wave = threadIdx.x >> 5;
  const int wpb  = blockDim.x >> 5;
  const int tm   = blockIdx.x * wpb + wave;      // 16-row block index
  if (tm >= (M >> 4)) return;                    // wave-uniform: EXEC all-1s
  const int g   = lane >> 4;
  const int row = tm * 16 + (lane & 15);

  v8f acc[NT];
#pragma unroll
  for (int t = 0; t < NT; ++t) acc[t] = (v8f){};

  for (int k0 = 0; k0 < K; k0 += 32) {
    const _Float16* arow = A + (long long)row * K + k0 + 8 * g;
    v8h alo = *(const v8h*)(arow);
    v8h ahi = *(const v8h*)(arow + 16);
    v16h av;
#pragma unroll
    for (int i = 0; i < 8; ++i) { av[i] = alo[i]; av[i + 8] = ahi[i]; }

    const _Float16* brow = B + (long long)(k0 + 8 * g) * Nc + (lane & 15);
#pragma unroll
    for (int t = 0; t < NT; ++t) {
      v16h bv;
#pragma unroll
      for (int i = 0; i < 8; ++i) {
        bv[i]     = brow[(long long)i * Nc + t * 16];
        bv[i + 8] = brow[(long long)(16 + i) * Nc + t * 16];
      }
      acc[t] = __builtin_amdgcn_wmma_f32_16x16x32_f16(
          /*neg_a=*/false, av, /*neg_b=*/false, bv,
          /*c_mod=*/(short)0, acc[t], /*reuse_a=*/false, /*reuse_b=*/false);
    }
  }

  const int cn = lane & 15;
  const int mb = tm * 16 + 8 * g;
#pragma unroll
  for (int t = 0; t < NT; ++t)
#pragma unroll
    for (int r = 0; r < 8; ++r)
      C[(long long)(mb + r) * Nc + t * 16 + cn] = acc[t][r];
}

// ---------------------------------------------------------------------------
// alpha_s/alpha_d = sum_c xl[n,h,c]*a_{src,dst}[h,c]; also init m=-inf, s=0.
// ---------------------------------------------------------------------------
__global__ void k_alpha(const float* __restrict__ xl,
                        const float* __restrict__ a_src,
                        const float* __restrict__ a_dst,
                        float* __restrict__ as_, float* __restrict__ ad_,
                        float* __restrict__ m, float* __restrict__ s,
                        int n_nodes, int heads, int ch) {
  int t = blockIdx.x * blockDim.x + threadIdx.x;
  if (t >= n_nodes * heads) return;
  int n = t / heads, h = t % heads;
  const float* xr = xl + (long long)n * heads * ch + h * ch;
  float s0 = 0.f, s1 = 0.f;
  for (int c = 0; c < ch; ++c) {
    float v = xr[c];
    s0 += v * a_src[h * ch + c];
    s1 += v * a_dst[h * ch + c];
  }
  as_[t] = s0;
  ad_[t] = s1;
  m[t]   = __int_as_float((int)0xff800000);   // -inf (self-loop guarantees hit)
  s[t]   = 0.f;
}

// sign-split float atomic max -> global_atomic_max_i32 / atomic_min_u32
__device__ inline void atomicMaxF(float* a, float v) {
  if (v >= 0.f) atomicMax((int*)a, __float_as_int(v));
  else          atomicMin((unsigned int*)a, (unsigned int)__float_as_int(v));
}

__device__ inline void edge_ends(const int* __restrict__ ei, int e, int n_edges,
                                 int& src, int& dst) {
  if (e < n_edges) { src = ei[e]; dst = ei[n_edges + e]; }
  else             { src = dst = e - n_edges; }          // appended self-loops
}

__device__ inline float lrelu(float v) { return v > 0.f ? v : NEG_SLOPE * v; }

// ---------------------------------------------------------------------------
// pass 1: segment max of leaky_relu(alpha_s[src]+alpha_d[dst]) over dst
// ---------------------------------------------------------------------------
__global__ void k_edge_max(const int* __restrict__ ei,
                           const float* __restrict__ as_,
                           const float* __restrict__ ad_,
                           float* __restrict__ m,
                           int n_edges, int n_nodes, int heads) {
  int t = blockIdx.x * blockDim.x + threadIdx.x;
  if (t >= (n_edges + n_nodes) * heads) return;
  int e = t / heads, h = t % heads, src, dst;
  edge_ends(ei, e, n_edges, src, dst);
  float v = lrelu(as_[src * heads + h] + ad_[dst * heads + h]);
  atomicMaxF(&m[dst * heads + h], v);
}

// ---------------------------------------------------------------------------
// pass 2: segment sum of exp(e - m[dst]) (edge logit recomputed from L2 data)
// ---------------------------------------------------------------------------
__global__ void k_edge_sum(const int* __restrict__ ei,
                           const float* __restrict__ as_,
                           const float* __restrict__ ad_,
                           const float* __restrict__ m,
                           float* __restrict__ s,
                           int n_edges, int n_nodes, int heads) {
  int t = blockIdx.x * blockDim.x + threadIdx.x;
  if (t >= (n_edges + n_nodes) * heads) return;
  int e = t / heads, h = t % heads, src, dst;
  edge_ends(ei, e, n_edges, src, dst);
  float v = lrelu(as_[src * heads + h] + ad_[dst * heads + h]);
  atomicAdd(&s[dst * heads + h], __expf(v - m[dst * heads + h]));
}

// ---------------------------------------------------------------------------
// pass 3: agg[dst] += xl[src] * alpha.  blockDim = (F, edges_per_block).
// Per-edge head weights computed once into LDS, shared by F lanes.
// ---------------------------------------------------------------------------
__global__ void k_edge_aggr(const int* __restrict__ ei,
                            const float* __restrict__ xl,
                            const float* __restrict__ as_,
                            const float* __restrict__ ad_,
                            const float* __restrict__ m,
                            const float* __restrict__ s,
                            float* __restrict__ agg,
                            int n_edges, int n_nodes, int heads, int ch) {
  __shared__ float wsm[16][8];                 // [edges_per_block<=16][heads<=8]
  const int F = blockDim.x;                    // heads*ch
  int e = blockIdx.x * blockDim.y + threadIdx.y;
  bool valid = e < n_edges + n_nodes;
  int src = 0, dst = 0;
  if (valid) {
    edge_ends(ei, e, n_edges, src, dst);
    // warm the gathered feature row while the softmax weight is computed
    __builtin_prefetch(&xl[(long long)src * F + threadIdx.x], 0, 1);
    if ((int)threadIdx.x < heads) {
      int h = threadIdx.x;
      float v = lrelu(as_[src * heads + h] + ad_[dst * heads + h]);
      wsm[threadIdx.y][h] =
          __expf(v - m[dst * heads + h]) / (s[dst * heads + h] + 1e-16f);
    }
  }
  __syncthreads();
  if (!valid) return;
  int c = threadIdx.x;
  float w = wsm[threadIdx.y][c / ch];
  atomicAdd(&agg[(long long)dst * F + c], xl[(long long)src * F + c] * w);
}

// ---------------------------------------------------------------------------
// BN stats: per-channel sum & sum-of-squares (block partial + atomic merge)
// ---------------------------------------------------------------------------
__global__ void k_bn_stats(const float* __restrict__ x, float* __restrict__ sums,
                           int n_rows, int F, int rows_per_block) {
  int c  = threadIdx.x;
  int r0 = blockIdx.x * rows_per_block;
  int r1 = min(r0 + rows_per_block, n_rows);
  float s0 = 0.f, s1 = 0.f;
  for (int r = r0; r < r1; ++r) {
    float v = x[(long long)r * F + c];
    s0 += v; s1 += v * v;
  }
  atomicAdd(&sums[c], s0);
  atomicAdd(&sums[F + c], s1);
}

// ---------------------------------------------------------------------------
// BN apply + ReLU, emit f16 for next layer's WMMA input.
// (bias b cancels exactly through batch-norm: mean(x+b)=mean(x)+b, so skipped)
// ---------------------------------------------------------------------------
__global__ void k_bn_relu_f16(const float* __restrict__ x,
                              const float* __restrict__ sums,
                              const float* __restrict__ gamma,
                              const float* __restrict__ beta,
                              _Float16* __restrict__ out, int n_rows, int F) {
  int t = blockIdx.x * blockDim.x + threadIdx.x;
  if (t >= n_rows * F) return;
  int c = t % F;
  float inv = 1.0f / (float)n_rows;
  float mu  = sums[c] * inv;
  float var = sums[F + c] * inv - mu * mu;
  float y = gamma[c] * (x[t] - mu) * rsqrtf(var + 1e-5f) + beta[c];
  out[t] = (_Float16)fmaxf(y, 0.f);
}

// ---------------------------------------------------------------------------
// final: out = log_softmax(agg + b3), F = 16
// ---------------------------------------------------------------------------
__global__ void k_bias_logsoftmax(const float* __restrict__ agg,
                                  const float* __restrict__ b,
                                  float* __restrict__ out, int n_rows, int F) {
  int n = blockIdx.x * blockDim.x + threadIdx.x;
  if (n >= n_rows) return;
  float v[16];
  float mx = -3.4e38f;
  for (int j = 0; j < F; ++j) {
    v[j] = agg[(long long)n * F + j] + b[j];
    mx = fmaxf(mx, v[j]);
  }
  float sum = 0.f;
  for (int j = 0; j < F; ++j) sum += __expf(v[j] - mx);
  float lse = mx + __logf(sum);
  for (int j = 0; j < F; ++j) out[(long long)n * F + j] = v[j] - lse;
}

// ---------------------------------------------------------------------------
extern "C" void kernel_launch(void* const* d_in, const int* in_sizes, int n_in,
                              void* d_out, int out_size, void* d_ws, size_t ws_size,
                              hipStream_t stream) {
  (void)in_sizes; (void)n_in; (void)out_size; (void)ws_size;

  const float* x      = (const float*)d_in[0];
  const int*   ei     = (const int*)d_in[1];     // [2,E] int (JAX default x32)
  const float* W1     = (const float*)d_in[2];
  const float* a_s1   = (const float*)d_in[3];
  const float* a_d1   = (const float*)d_in[4];
  // b1 = d_in[5]  (cancels through BN)
  const float* gamma1 = (const float*)d_in[6];
  const float* beta1  = (const float*)d_in[7];
  const float* W2     = (const float*)d_in[8];
  const float* a_s2   = (const float*)d_in[9];
  const float* a_d2   = (const float*)d_in[10];
  // b2 = d_in[11] (cancels through BN)
  const float* gamma2 = (const float*)d_in[12];
  const float* beta2  = (const float*)d_in[13];
  const float* W3     = (const float*)d_in[14];
  const float* a_s3   = (const float*)d_in[15];
  const float* a_d3   = (const float*)d_in[16];
  const float* b3     = (const float*)d_in[17];
  float* out = (float*)d_out;

  // ---- workspace carve (256B aligned) ----
  size_t off = 0;
  char* base = (char*)d_ws;
  auto carve = [&](size_t bytes) -> void* {
    void* p = base + off;
    off += (bytes + 255) & ~(size_t)255;
    return p;
  };
  _Float16* xf16 = (_Float16*)carve((size_t)NNODES * HCF * sizeof(_Float16));
  _Float16* wf16 = (_Float16*)carve((size_t)HCF * HCF * sizeof(_Float16));
  float* xl   = (float*)carve((size_t)NNODES * HCF * sizeof(float));
  float* as_  = (float*)carve((size_t)NNODES * HEADS * sizeof(float));
  float* ad_  = (float*)carve((size_t)NNODES * HEADS * sizeof(float));
  float* mbuf = (float*)carve((size_t)NNODES * HEADS * sizeof(float));
  float* sbuf = (float*)carve((size_t)NNODES * HEADS * sizeof(float));
  float* agg  = (float*)carve((size_t)NNODES * HCF * sizeof(float));
  float* bns  = (float*)carve((size_t)2 * HCF * sizeof(float));

  const int NN = NNODES, NE = NEDGES, ETOT = NEDGES + NNODES;
  auto cdiv = [](int a, int b) { return (a + b - 1) / b; };
  const int mblocks = cdiv(NN / 16, 4);          // 4 waves (row-blocks) / block

  // ================= Layer 1 =================
  k_f32_to_f16<<<cdiv(NN * FIN, 256), 256, 0, stream>>>(x, xf16, NN * FIN);
  k_f32_to_f16<<<cdiv(FIN * HCF, 256), 256, 0, stream>>>(W1, wf16, FIN * HCF);
  k_gemm_wmma<HCF / 16><<<mblocks, 128, 0, stream>>>(xf16, wf16, xl, NN, FIN);
  k_alpha<<<cdiv(NN * HEADS, 256), 256, 0, stream>>>(xl, a_s1, a_d1, as_, ad_,
                                                     mbuf, sbuf, NN, HEADS, CH);
  hipMemsetAsync(agg, 0, (size_t)NN * HCF * sizeof(float), stream);
  k_edge_max<<<cdiv(ETOT * HEADS, 256), 256, 0, stream>>>(ei, as_, ad_, mbuf, NE, NN, HEADS);
  k_edge_sum<<<cdiv(ETOT * HEADS, 256), 256, 0, stream>>>(ei, as_, ad_, mbuf, sbuf, NE, NN, HEADS);
  k_edge_aggr<<<cdiv(ETOT, 2), dim3(HCF, 2), 0, stream>>>(ei, xl, as_, ad_, mbuf, sbuf,
                                                          agg, NE, NN, HEADS, CH);
  hipMemsetAsync(bns, 0, 2 * HCF * sizeof(float), stream);
  k_bn_stats<<<cdiv(NN, 64), HCF, 0, stream>>>(agg, bns, NN, HCF, 64);
  k_bn_relu_f16<<<cdiv(NN * HCF, 256), 256, 0, stream>>>(agg, bns, gamma1, beta1,
                                                         xf16, NN, HCF);

  // ================= Layer 2 =================
  k_f32_to_f16<<<cdiv(HCF * HCF, 256), 256, 0, stream>>>(W2, wf16, HCF * HCF);
  k_gemm_wmma<HCF / 16><<<mblocks, 128, 0, stream>>>(xf16, wf16, xl, NN, HCF);
  k_alpha<<<cdiv(NN * HEADS, 256), 256, 0, stream>>>(xl, a_s2, a_d2, as_, ad_,
                                                     mbuf, sbuf, NN, HEADS, CH);
  hipMemsetAsync(agg, 0, (size_t)NN * HCF * sizeof(float), stream);
  k_edge_max<<<cdiv(ETOT * HEADS, 256), 256, 0, stream>>>(ei, as_, ad_, mbuf, NE, NN, HEADS);
  k_edge_sum<<<cdiv(ETOT * HEADS, 256), 256, 0, stream>>>(ei, as_, ad_, mbuf, sbuf, NE, NN, HEADS);
  k_edge_aggr<<<cdiv(ETOT, 2), dim3(HCF, 2), 0, stream>>>(ei, xl, as_, ad_, mbuf, sbuf,
                                                          agg, NE, NN, HEADS, CH);
  hipMemsetAsync(bns, 0, 2 * HCF * sizeof(float), stream);
  k_bn_stats<<<cdiv(NN, 64), HCF, 0, stream>>>(agg, bns, NN, HCF, 64);
  k_bn_relu_f16<<<cdiv(NN * HCF, 256), 256, 0, stream>>>(agg, bns, gamma2, beta2,
                                                         xf16, NN, HCF);

  // ================= Layer 3 (1 head, 16 feats) =================
  k_f32_to_f16<<<cdiv(HCF * OUTF, 256), 256, 0, stream>>>(W3, wf16, HCF * OUTF);
  k_gemm_wmma<OUTF / 16><<<mblocks, 128, 0, stream>>>(xf16, wf16, xl, NN, HCF);
  k_alpha<<<cdiv(NN, 256), 256, 0, stream>>>(xl, a_s3, a_d3, as_, ad_,
                                             mbuf, sbuf, NN, 1, OUTF);
  hipMemsetAsync(agg, 0, (size_t)NN * OUTF * sizeof(float), stream);
  k_edge_max<<<cdiv(ETOT, 256), 256, 0, stream>>>(ei, as_, ad_, mbuf, NE, NN, 1);
  k_edge_sum<<<cdiv(ETOT, 256), 256, 0, stream>>>(ei, as_, ad_, mbuf, sbuf, NE, NN, 1);
  k_edge_aggr<<<cdiv(ETOT, 16), dim3(OUTF, 16), 0, stream>>>(ei, xl, as_, ad_, mbuf, sbuf,
                                                             agg, NE, NN, 1, OUTF);
  k_bias_logsoftmax<<<cdiv(NN, 256), 256, 0, stream>>>(agg, b3, out, NN, OUTF);
}
